// LSTM_3444563771996
// MI455X (gfx1250) — matrix operand order repaired and checked by reading, not compiled, not run on hardware
//
#include <hip/hip_runtime.h>
#include <hip/hip_bf16.h>

// ---------------------------------------------------------------------------
// LSTM fused scan for MI455X (gfx1250), wave32 + WMMA bf16.
//   I=28, H=64, T=128, B=4096, OUT=10
// Pass 1: x f32 -> bf16, K padded 28->32; pad col 28 carries the constant 1.0
//         so the combined bias rides inside W_ih's pad column (acc C = inline 0).
// Pass 2: one WG per 16-row batch tile; 4 waves; wave w owns hidden cols
//         16w..16w+15 (gate tiles w, w+4, w+8, w+12 = i,f,g,o).
//         Per step: 12x v_wmma_f32_16x16x32_bf16, hardware v_tanh_f32
//         activations, h exchanged via double-buffered LDS, 1 barrier,
//         x loads software-pipelined one step ahead + prefetch t+2.
// ---------------------------------------------------------------------------

typedef __bf16 bf16_t;
typedef __bf16 v16bf __attribute__((ext_vector_type(16)));
typedef __bf16 v8bf  __attribute__((ext_vector_type(8)));
typedef float  v8f   __attribute__((ext_vector_type(8)));

#define I_DIM   28
#define BIAS_K  28        // pad column that carries 1.0 / bias
#define KPAD    32
#define H_DIM   64
#define T_STEPS 128
#define B_DIM   4096
#define OUT_DIM 10
#define G4H     256       // 4*H

static __device__ inline v16bf cat8(v8bf lo, v8bf hi) {
  union { v16bf v; v8bf h[2]; } u;
  u.h[0] = lo; u.h[1] = hi;
  return u.v;
}

// Branch-free activations on the TRANS32 pipe (co-executes with WMMA).
static __device__ inline float fast_tanh(float x) {
#if __has_builtin(__builtin_amdgcn_tanhf)
  return __builtin_amdgcn_tanhf(x);              // v_tanh_f32
#else
  float t = __builtin_amdgcn_exp2f(x * 2.88539008177792681f);
  return 1.0f - 2.0f * __builtin_amdgcn_rcpf(t + 1.0f);
#endif
}

static __device__ inline float fast_sigmoid(float x) {
  return __builtin_fmaf(0.5f, fast_tanh(0.5f * x), 0.5f);
}

// ---------------------------------------------------------------------------
// Pass 1: x [B][T][28] f32 -> x_pre [B/16][T][16][32] bf16
//         (k<28: data, k==28: 1.0 for bias, k>28: 0)
// ---------------------------------------------------------------------------
__global__ __launch_bounds__(256) void lstm_xpre_kernel(
    const float* __restrict__ x, bf16_t* __restrict__ x_pre) {
  size_t idx = (size_t)blockIdx.x * 256 + threadIdx.x;
  const size_t total = (size_t)B_DIM * T_STEPS * KPAD;
  if (idx >= total) return;
  int    k  = (int)(idx & (KPAD - 1));
  size_t bt = idx >> 5;                // b*T + t
  int    t  = (int)(bt & (T_STEPS - 1));
  int    b  = (int)(bt >> 7);
  float v = (k < I_DIM) ? x[bt * I_DIM + k] : ((k == BIAS_K) ? 1.0f : 0.0f);
  size_t dst = ((((size_t)(b >> 4)) * T_STEPS + t) * 16 + (b & 15)) * KPAD + k;
  x_pre[dst] = (bf16_t)v;
}

// ---------------------------------------------------------------------------
// Pass 2: fused LSTM scan + output projection
// ---------------------------------------------------------------------------
__global__ __launch_bounds__(128) void lstm_scan_kernel(
    const bf16_t* __restrict__ x_pre,   // [256][T][16][32] bf16
    const float*  __restrict__ W_ih,    // [256][28]
    const float*  __restrict__ W_hh,    // [256][64]
    const float*  __restrict__ b_ih,    // [256]
    const float*  __restrict__ b_hh,    // [256]
    const float*  __restrict__ W_out,   // [10][64]
    const float*  __restrict__ b_out,   // [10]
    float*        __restrict__ out)     // [B][10]
{
  // B-fragment-friendly weight staging (lane = gate col N, elems = contiguous K)
  __shared__ __align__(32) bf16_t s_wih[G4H][KPAD];    // 16 KB (col 28 = bias)
  __shared__ __align__(32) bf16_t s_whh[G4H][H_DIM];   // 32 KB
  __shared__ __align__(32) bf16_t s_h[2][16][H_DIM];   // 4 KB, double buffer

  const int tid   = threadIdx.x;
  const int wave  = tid >> 5;        // 0..3, owns hidden cols 16*wave..+15
  const int lane  = tid & 31;
  const int half  = lane >> 4;       // A/B: K-half selector; D: M half
  const int lrow  = lane & 15;       // A row M; B/D col N
  const int mtile = blockIdx.x;      // batch tile (16 rows)

  // ---- stage weights f32 -> bf16 into LDS (bias folded into W_ih col 28) ----
  for (int idx = tid; idx < G4H * KPAD; idx += 128) {
    int n = idx >> 5, k = idx & (KPAD - 1);
    float v = (k < I_DIM) ? W_ih[n * I_DIM + k]
                          : ((k == BIAS_K) ? (b_ih[n] + b_hh[n]) : 0.0f);
    s_wih[n][k] = (bf16_t)v;
  }
  for (int idx = tid; idx < G4H * H_DIM; idx += 128)
    s_whh[idx >> 6][idx & 63] = (bf16_t)W_hh[idx];
  for (int idx = tid; idx < 16 * H_DIM; idx += 128)
    s_h[0][idx >> 6][idx & 63] = (bf16_t)0.0f;     // h0 = 0
  __syncthreads();

  // ---- per-wave gate tiles: i,f,g,o ----
  const int ni = wave, nf = wave + 4, ng = wave + 8, no_ = wave + 12;

  // B fragments held in registers for the whole scan.
  // B (32x16): elem j -> K = 16*half + j (contiguous)
  v16bf Bih_i = *(const v16bf*)&s_wih[16 * ni  + lrow][16 * half];
  v16bf Bih_f = *(const v16bf*)&s_wih[16 * nf  + lrow][16 * half];
  v16bf Bih_g = *(const v16bf*)&s_wih[16 * ng  + lrow][16 * half];
  v16bf Bih_o = *(const v16bf*)&s_wih[16 * no_ + lrow][16 * half];
  v16bf Bhh_i0 = *(const v16bf*)&s_whh[16 * ni  + lrow][16 * half];
  v16bf Bhh_i1 = *(const v16bf*)&s_whh[16 * ni  + lrow][32 + 16 * half];
  v16bf Bhh_f0 = *(const v16bf*)&s_whh[16 * nf  + lrow][16 * half];
  v16bf Bhh_f1 = *(const v16bf*)&s_whh[16 * nf  + lrow][32 + 16 * half];
  v16bf Bhh_g0 = *(const v16bf*)&s_whh[16 * ng  + lrow][16 * half];
  v16bf Bhh_g1 = *(const v16bf*)&s_whh[16 * ng  + lrow][32 + 16 * half];
  v16bf Bhh_o0 = *(const v16bf*)&s_whh[16 * no_ + lrow][16 * half];
  v16bf Bhh_o1 = *(const v16bf*)&s_whh[16 * no_ + lrow][32 + 16 * half];

  v8f c_state;
#pragma unroll
  for (int e = 0; e < 8; ++e) c_state[e] = 0.0f;

  const bf16_t* xbase =
      x_pre + (((size_t)mtile * T_STEPS) * 16 + lrow) * KPAD;

  // Software pipeline: xA for step t is loaded during step t-1.
  v16bf xA = cat8(*(const v8bf*)&xbase[16 * half],
                  *(const v8bf*)&xbase[8 + 16 * half]);

  int cur = 0;
  for (int t = 0; t < T_STEPS; ++t) {
    // ---- issue next step's x loads early (consumed after the barrier) ----
    int tn = (t + 1 < T_STEPS) ? (t + 1) : t;
    const bf16_t* xn = xbase + (size_t)tn * (16 * KPAD);
    v16bf xA_next = cat8(*(const v8bf*)&xn[16 * half],
                         *(const v8bf*)&xn[8 + 16 * half]);
    __builtin_prefetch(xn + 16 * KPAD, 0, 2);       // t+2 -> near cache

    // ---- A fragments of h (16x64 -> two 16x32 chunks) from LDS ----
    const bf16_t* hrow = &s_h[cur][lrow][0];
    v16bf hA0 = cat8(*(const v8bf*)&hrow[16 * half],
                     *(const v8bf*)&hrow[8 + 16 * half]);
    v16bf hA1 = cat8(*(const v8bf*)&hrow[32 + 16 * half],
                     *(const v8bf*)&hrow[40 + 16 * half]);

    // ---- gates = [x|1]@[Wih|b]^T + h@Whh^T  (12 WMMAs, C starts inline 0) ----
    v8f zero = {};
    v8f ai, af, ag, ao;
    ai = __builtin_amdgcn_wmma_f32_16x16x32_bf16(false, xA,  false, Bih_i,  (short)0, zero, false, false);
    af = __builtin_amdgcn_wmma_f32_16x16x32_bf16(false, xA,  false, Bih_f,  (short)0, zero, false, false);
    ag = __builtin_amdgcn_wmma_f32_16x16x32_bf16(false, xA,  false, Bih_g,  (short)0, zero, false, false);
    ao = __builtin_amdgcn_wmma_f32_16x16x32_bf16(false, xA,  false, Bih_o,  (short)0, zero, false, false);
    ai = __builtin_amdgcn_wmma_f32_16x16x32_bf16(false, hA0, false, Bhh_i0, (short)0, ai, false, false);
    af = __builtin_amdgcn_wmma_f32_16x16x32_bf16(false, hA0, false, Bhh_f0, (short)0, af, false, false);
    ag = __builtin_amdgcn_wmma_f32_16x16x32_bf16(false, hA0, false, Bhh_g0, (short)0, ag, false, false);
    ao = __builtin_amdgcn_wmma_f32_16x16x32_bf16(false, hA0, false, Bhh_o0, (short)0, ao, false, false);
    ai = __builtin_amdgcn_wmma_f32_16x16x32_bf16(false, hA1, false, Bhh_i1, (short)0, ai, false, false);
    af = __builtin_amdgcn_wmma_f32_16x16x32_bf16(false, hA1, false, Bhh_f1, (short)0, af, false, false);
    ag = __builtin_amdgcn_wmma_f32_16x16x32_bf16(false, hA1, false, Bhh_g1, (short)0, ag, false, false);
    ao = __builtin_amdgcn_wmma_f32_16x16x32_bf16(false, hA1, false, Bhh_o1, (short)0, ao, false, false);

    // ---- branch-free activations + state update (D layout) ----
    int nxt = cur ^ 1;
#pragma unroll
    for (int e = 0; e < 8; ++e) {
      float iv = fast_sigmoid(ai[e]);
      float fv = fast_sigmoid(af[e]);
      float gv = fast_tanh(ag[e]);
      float ov = fast_sigmoid(ao[e]);
      float cv = __builtin_fmaf(fv, c_state[e], iv * gv);
      c_state[e] = cv;
      float hv = ov * fast_tanh(cv);
      // elem e -> (batch row e + 8*half, hidden col 16*wave + lrow)
      s_h[nxt][e + 8 * half][16 * wave + lrow] = (bf16_t)hv;
    }
    __syncthreads();
    cur = nxt;
    xA = xA_next;
  }

  // ---- final projection: out = h_T @ W_out^T + b_out ----
  for (int idx = tid; idx < 16 * OUT_DIM; idx += 128) {
    int bl = idx / OUT_DIM;
    int o  = idx % OUT_DIM;
    float acc = b_out[o];
#pragma unroll 16
    for (int k = 0; k < H_DIM; ++k)
      acc += (float)s_h[cur][bl][k] * W_out[o * H_DIM + k];
    out[((size_t)mtile * 16 + bl) * OUT_DIM + o] = acc;
  }
}

// ---------------------------------------------------------------------------
extern "C" void kernel_launch(void* const* d_in, const int* in_sizes, int n_in,
                              void* d_out, int out_size, void* d_ws, size_t ws_size,
                              hipStream_t stream) {
  const float* x     = (const float*)d_in[0];
  const float* W_ih  = (const float*)d_in[1];
  const float* W_hh  = (const float*)d_in[2];
  const float* b_ih  = (const float*)d_in[3];
  const float* b_hh  = (const float*)d_in[4];
  const float* W_out = (const float*)d_in[5];
  const float* b_out = (const float*)d_in[6];
  float* out = (float*)d_out;

  // Workspace: bf16 x staging, B*T*32*2 = 32 MB (assumed <= ws_size).
  bf16_t* x_pre = (bf16_t*)d_ws;

  const size_t xpre_elems = (size_t)B_DIM * T_STEPS * KPAD;
  const int xpre_blocks = (int)((xpre_elems + 255) / 256);
  lstm_xpre_kernel<<<xpre_blocks, 256, 0, stream>>>(x, x_pre);

  lstm_scan_kernel<<<B_DIM / 16, 128, 0, stream>>>(
      x_pre, W_ih, W_hh, b_ih, b_hh, W_out, b_out, out);
}
